// GraphToVectorGNN_24867860644045
// MI455X (gfx1250) — matrix operand order repaired and checked by reading, not compiled, not run on hardware
//
#include <hip/hip_runtime.h>

typedef __attribute__((ext_vector_type(16))) __bf16 v16bf;
typedef __attribute__((ext_vector_type(8)))  float  v8f;

union BfFrag { v16bf v; unsigned short s[16]; unsigned int u[8]; uint4 q[2]; };

#define WMMA_D 128

// ---------------- elementwise / utility kernels ----------------

__global__ void fzero_kernel(float* __restrict__ p, int n) {
    int i = blockIdx.x * blockDim.x + threadIdx.x;
    if (i < n) p[i] = 0.0f;
}

__device__ inline unsigned short f2bf_rne(float f) {
    unsigned u = __float_as_uint(f);
    u += 0x7FFFu + ((u >> 16) & 1u);   // round-to-nearest-even bf16
    return (unsigned short)(u >> 16);
}

__global__ void f2bf_kernel(const float* __restrict__ in, unsigned short* __restrict__ out, int n) {
    int i = blockIdx.x * blockDim.x + threadIdx.x;
    if (i >= n) return;
    out[i] = f2bf_rne(in[i]);
}

// Pack weight W[K,Nc] (fp32 row-major) into WMMA B-fragment order (bf16):
// out[((kt*ntiles + tn)*32 + lane)*16 + j]; lane -> (half=lane>>4, col=tn*16+(lane&15));
// j walks K contiguously: K = kt*32 + half*16 + j.  16 contiguous shorts per lane.
__global__ void packB_kernel(const float* __restrict__ W, unsigned short* __restrict__ out,
                             int K, int Nc) {
    int tid = blockIdx.x * blockDim.x + threadIdx.x;
    if (tid >= K * Nc) return;
    int j    = tid & 15;
    int lane = (tid >> 4) & 31;
    int ft   = tid >> 9;               // kt*ntiles + tn
    int ntiles = Nc >> 4;
    int tn = ft % ntiles;
    int kt = ft / ntiles;
    int half = lane >> 4;
    int col  = tn * 16 + (lane & 15);
    int k    = kt * 32 + half * 16 + j;
    out[tid] = f2bf_rne(W[(size_t)k * Nc + col]);
}

__global__ void deg_kernel(const int* __restrict__ dst, float* __restrict__ deg, int E) {
    int e = blockIdx.x * blockDim.x + threadIdx.x;
    if (e < E) atomicAdd(&deg[dst[e]], 1.0f);
}

__global__ void dinv_kernel(float* __restrict__ d, int n) {
    int i = blockIdx.x * blockDim.x + threadIdx.x;
    if (i < n) d[i] = rsqrtf(d[i] + 1.0f);
}

// ---------------- WMMA bf16 GEMM ----------------
// C[M,Nc] = A[M,K] @ B[K,Nc] (+bias)(tanh?).  A: natural row-major bf16.
// Bp: packed fragment layout (packB_kernel).  One wave = 16 rows x NT col-tiles.
// NT is a compile-time constant -> branch-free inner loop, EXEC provably all-ones at WMMA.

#define WMMA_BF16(a, b, c) __builtin_amdgcn_wmma_f32_16x16x32_bf16( \
    false, (a), false, (b), (short)0, (c), false, false)

template <int NT>
__global__ void wmma_gemm_bf16(const unsigned short* __restrict__ A,
                               const unsigned short* __restrict__ Bp,
                               const float* __restrict__ bias,
                               float* __restrict__ C,
                               int M, int K, int Nc, int act_tanh) {
    int ntiles  = Nc >> 4;
    int nchunks = ntiles / NT;            // ntiles is a multiple of NT by construction
    int mtiles  = (M + 15) >> 4;
    int wid = (blockIdx.x * blockDim.x + threadIdx.x) >> 5;
    if (wid >= mtiles * nchunks) return;  // whole-wave exit
    int tm  = wid / nchunks;
    int tn0 = (wid % nchunks) * NT;
    int lane = threadIdx.x & 31;
    int half = lane >> 4;
    int l15  = lane & 15;
    int row  = tm * 16 + l15; if (row >= M) row = M - 1;

    v8f acc[NT];
#pragma unroll
    for (int t = 0; t < NT; ++t) acc[t] = (v8f){};

    const size_t bstride = (size_t)ntiles * 32 * 16;   // shorts per K-step of packed B
    const unsigned short* Bk = Bp + ((size_t)tn0 * 32 + lane) * 16;

    for (int k0 = 0; k0 < K; k0 += 32, Bk += bstride) {
        BfFrag a;
        const unsigned short* Arow = A + (size_t)row * K + k0;
        a.q[0] = *(const uint4*)(Arow + half * 16);       // K = h*16 .. h*16+7
        a.q[1] = *(const uint4*)(Arow + half * 16 + 8);   // K = h*16+8 ..
        __builtin_prefetch(Bk + bstride, 0, 3);           // next K-step packed B

#pragma unroll
        for (int t = 0; t < NT; ++t) {
            BfFrag b;
            b.q[0] = *(const uint4*)(Bk + t * 32 * 16);
            b.q[1] = *(const uint4*)(Bk + t * 32 * 16 + 8);
            acc[t] = WMMA_BF16(a.v, b.v, acc[t]);
        }
    }

#pragma unroll
    for (int t = 0; t < NT; ++t) {
        int col = (tn0 + t) * 16 + l15;
        float bv = bias ? bias[col] : 0.0f;
#pragma unroll
        for (int v = 0; v < 8; ++v) {
            int r = tm * 16 + half * 8 + v;               // C/D layout
            if (r < M) {
                float val = acc[t][v] + bv;
                if (act_tanh) val = tanhf(val);
                C[(size_t)r * Nc + col] = val;
            }
        }
    }
}

// ---------------- GCN message passing (scatter-add through L2 atomics) ----------------

__global__ void edge_msg_kernel(const int* __restrict__ src, const int* __restrict__ dst,
                                const float* __restrict__ dinv,
                                const float* __restrict__ h, float* __restrict__ agg, int E) {
    int e = (blockIdx.x * blockDim.x + threadIdx.x) >> 5;
    if (e >= E) return;
    int lane = threadIdx.x & 31;
    int s = src[e], d = dst[e];
    float coef = dinv[s] * dinv[d];
    const float4* hv = (const float4*)(h + (size_t)s * WMMA_D);
    float4 m = hv[lane];
    float* out = agg + (size_t)d * WMMA_D + lane * 4;
    atomicAdd(out + 0, m.x * coef);
    atomicAdd(out + 1, m.y * coef);
    atomicAdd(out + 2, m.z * coef);
    atomicAdd(out + 3, m.w * coef);
}

__global__ void gcn_finalize_kernel(float* __restrict__ agg, const float* __restrict__ h,
                                    const float* __restrict__ dinv,
                                    const float* __restrict__ bias, int total) {
    int i = blockIdx.x * blockDim.x + threadIdx.x;
    if (i >= total) return;
    int node = i >> 7, f = i & (WMMA_D - 1);
    float di = dinv[node];
    agg[i] = agg[i] + h[i] * di * di + bias[f];
}

// ---------------- row LayerNorm (one wave per row), optional tanh before/after ----------------

__global__ void rowln_kernel(float* __restrict__ X, const float* __restrict__ w,
                             const float* __restrict__ b, int rows, int D,
                             int pre_tanh, int post_tanh) {
    int row = (blockIdx.x * blockDim.x + threadIdx.x) >> 5;
    if (row >= rows) return;
    int lane = threadIdx.x & 31;
    float vals[8];
    int cnt = D >> 5;                      // D in {128,256}
    float s = 0.0f;
    for (int i = 0; i < cnt; ++i) {
        float v = X[(size_t)row * D + lane + (i << 5)];
        if (pre_tanh) v = tanhf(v);
        vals[i] = v; s += v;
    }
    for (int o = 16; o > 0; o >>= 1) s += __shfl_xor(s, o, 32);
    float mu = s / (float)D;
    float q = 0.0f;
    for (int i = 0; i < cnt; ++i) { float d = vals[i] - mu; q += d * d; }
    for (int o = 16; o > 0; o >>= 1) q += __shfl_xor(q, o, 32);
    float inv = rsqrtf(q / (float)D + 1e-5f);
    for (int i = 0; i < cnt; ++i) {
        int j = lane + (i << 5);
        float y = (vals[i] - mu) * inv * w[j] + b[j];
        if (post_tanh) y = tanhf(y);
        X[(size_t)row * D + j] = y;
    }
}

// ---------------- gate tail (K=16 dot), segment softmax, attention pooling ----------------

__global__ void gate_final_kernel(const float* __restrict__ g16, const float* __restrict__ W,
                                  const float* __restrict__ b, float* __restrict__ gate, int n) {
    int i = blockIdx.x * blockDim.x + threadIdx.x;
    if (i >= n) return;
    float s = b[0];
#pragma unroll
    for (int k = 0; k < 16; ++k) s += g16[(size_t)i * 16 + k] * W[k];
    gate[i] = s;
}

__device__ inline unsigned f_enc(float f) {
    unsigned u = __float_as_uint(f);
    return (u & 0x80000000u) ? ~u : (u | 0x80000000u);
}
__device__ inline float f_dec(unsigned u) {
    unsigned b = (u & 0x80000000u) ? (u & 0x7FFFFFFFu) : ~u;
    return __uint_as_float(b);
}

__global__ void seg_max_kernel(const float* __restrict__ gate, const int* __restrict__ batch,
                               unsigned* __restrict__ gmax, int n) {
    int i = blockIdx.x * blockDim.x + threadIdx.x;
    if (i >= n) return;
    atomicMax(&gmax[batch[i]], f_enc(gate[i]));
}

__global__ void seg_exp_kernel(float* __restrict__ gate, const int* __restrict__ batch,
                               const unsigned* __restrict__ gmax, float* __restrict__ den, int n) {
    int i = blockIdx.x * blockDim.x + threadIdx.x;
    if (i >= n) return;
    float e = __expf(gate[i] - f_dec(gmax[batch[i]]));
    gate[i] = e;
    atomicAdd(&den[batch[i]], e);
}

__global__ void pool_kernel(const float* __restrict__ H, const float* __restrict__ e,
                            const float* __restrict__ den, const int* __restrict__ batch,
                            float* __restrict__ pooled, int n) {
    int node = (blockIdx.x * blockDim.x + threadIdx.x) >> 5;
    if (node >= n) return;
    int lane = threadIdx.x & 31;
    int g = batch[node];
    float dn = den[g]; dn = dn > 0.0f ? dn : 1.0f;
    float a = e[node] / dn;
    const float4* hv = (const float4*)(H + (size_t)node * WMMA_D);
    float4 m = hv[lane];
    float* out = pooled + (size_t)g * WMMA_D + lane * 4;
    atomicAdd(out + 0, m.x * a);
    atomicAdd(out + 1, m.y * a);
    atomicAdd(out + 2, m.z * a);
    atomicAdd(out + 3, m.w * a);
}

// ---------------- host side ----------------

static inline int grids(long long threads, int block) { return (int)((threads + block - 1) / block); }

extern "C" void kernel_launch(void* const* d_in, const int* in_sizes, int n_in,
                              void* d_out, int out_size, void* d_ws, size_t ws_size,
                              hipStream_t stream) {
    const float* x     = (const float*)d_in[0];
    const int*   ei    = (const int*)d_in[1];
    const int*   batch = (const int*)d_in[2];
    const float* W1 = (const float*)d_in[3],  *b1 = (const float*)d_in[4];
    const float* ln1w = (const float*)d_in[5], *ln1b = (const float*)d_in[6];
    const float* W2 = (const float*)d_in[7],  *b2 = (const float*)d_in[8];
    const float* ln2w = (const float*)d_in[9], *ln2b = (const float*)d_in[10];
    const float* ppw = (const float*)d_in[11], *ppb = (const float*)d_in[12];
    const float* g1W = (const float*)d_in[13], *g1b = (const float*)d_in[14];
    const float* g2W = (const float*)d_in[15], *g2b = (const float*)d_in[16];
    const float* g3W = (const float*)d_in[17], *g3b = (const float*)d_in[18];
    const float* g4W = (const float*)d_in[19], *g4b = (const float*)d_in[20];
    const float* m1W = (const float*)d_in[21], *m1b = (const float*)d_in[22];
    const float* m1lnw = (const float*)d_in[23], *m1lnb = (const float*)d_in[24];
    const float* m2W = (const float*)d_in[25], *m2b = (const float*)d_in[26];
    const float* m2lnw = (const float*)d_in[27], *m2lnb = (const float*)d_in[28];
    const float* m3W = (const float*)d_in[29], *m3b = (const float*)d_in[30];

    const int N = in_sizes[2];
    const int E = in_sizes[1] / 2;
    const int G = 512;
    const int D = WMMA_D;
    const int* src = ei;
    const int* dst = ei + E;

    // workspace carving
    char* wsb = (char*)d_ws;
    size_t off = 0;
    auto alloc = [&](size_t bytes) -> void* {
        void* p = wsb + off;
        off = (off + bytes + 255) & ~(size_t)255;
        return p;
    };
    float* dinv   = (float*)alloc((size_t)N * 4);
    float* bufA   = (float*)alloc((size_t)N * D * 4);
    float* bufB   = (float*)alloc((size_t)N * D * 4);
    unsigned short* bfb = (unsigned short*)alloc((size_t)N * D * 2);
    float* gate   = (float*)alloc((size_t)N * 4);
    unsigned* gmaxb = (unsigned*)alloc((size_t)G * 4);
    float* den    = (float*)alloc((size_t)G * 4);
    float* pooled = (float*)alloc((size_t)G * D * 4);
    float* z1     = (float*)alloc((size_t)G * 256 * 4);
    float* z2     = (float*)alloc((size_t)G * 128 * 4);
    unsigned short* bfs = (unsigned short*)alloc((size_t)G * 256 * 2);
    unsigned short* w1b = (unsigned short*)alloc(128 * 128 * 2);
    unsigned short* w2b = (unsigned short*)alloc(128 * 128 * 2);
    unsigned short* g1wb = (unsigned short*)alloc(128 * 128 * 2);
    unsigned short* g2wb = (unsigned short*)alloc(128 * 64 * 2);
    unsigned short* g3wb = (unsigned short*)alloc(64 * 16 * 2);
    unsigned short* m1wb = (unsigned short*)alloc(128 * 256 * 2);
    unsigned short* m2wb = (unsigned short*)alloc(256 * 128 * 2);
    unsigned short* m3wb = (unsigned short*)alloc(128 * 64 * 2);

    const int BS = 256;
    auto gemm = [&](const unsigned short* A, const unsigned short* B, const float* bias,
                    float* C, int M, int K, int Nc, int act) {
        int ntiles = Nc / 16;
        if ((ntiles & 3) == 0) {
            long long waves = (long long)((M + 15) / 16) * (ntiles / 4);
            wmma_gemm_bf16<4><<<grids(waves * 32, BS), BS, 0, stream>>>(A, B, bias, C, M, K, Nc, act);
        } else {
            long long waves = (long long)((M + 15) / 16) * ntiles;
            wmma_gemm_bf16<1><<<grids(waves * 32, BS), BS, 0, stream>>>(A, B, bias, C, M, K, Nc, act);
        }
    };
    auto packW = [&](const float* W, unsigned short* out, int K, int Nc) {
        packB_kernel<<<grids((long long)K * Nc, BS), BS, 0, stream>>>(W, out, K, Nc);
    };
    auto cvt = [&](const float* in, unsigned short* out, long long n) {
        f2bf_kernel<<<grids(n, BS), BS, 0, stream>>>(in, out, (int)n);
    };
    auto zero = [&](float* p, long long n) {
        fzero_kernel<<<grids(n, BS), BS, 0, stream>>>(p, (int)n);
    };
    auto ln = [&](float* X, const float* w, const float* b, int rows, int Dw, int pre, int post) {
        rowln_kernel<<<grids((long long)rows * 32, BS), BS, 0, stream>>>(X, w, b, rows, Dw, pre, post);
    };

    // degree -> dinv
    zero(dinv, N);
    deg_kernel<<<grids(E, BS), BS, 0, stream>>>(dst, dinv, E);
    dinv_kernel<<<grids(N, BS), BS, 0, stream>>>(dinv, N);

    // weight conversions into packed WMMA fragment order
    packW(W1, w1b, 128, 128);   packW(W2, w2b, 128, 128);
    packW(g1W, g1wb, 128, 128); packW(g2W, g2wb, 128, 64); packW(g3W, g3wb, 64, 16);
    packW(m1W, m1wb, 128, 256); packW(m2W, m2wb, 256, 128); packW(m3W, m3wb, 128, 64);

    // GCN layer 1
    cvt(x, bfb, (long long)N * D);
    gemm(bfb, w1b, nullptr, bufA, N, 128, 128, 0);
    zero(bufB, (long long)N * D);
    edge_msg_kernel<<<grids((long long)E * 32, BS), BS, 0, stream>>>(src, dst, dinv, bufA, bufB, E);
    gcn_finalize_kernel<<<grids((long long)N * D, BS), BS, 0, stream>>>(bufB, bufA, dinv, b1, N * D);
    ln(bufB, ln1w, ln1b, N, 128, 1, 0);

    // GCN layer 2
    cvt(bufB, bfb, (long long)N * D);
    gemm(bfb, w2b, nullptr, bufA, N, 128, 128, 0);
    zero(bufB, (long long)N * D);
    edge_msg_kernel<<<grids((long long)E * 32, BS), BS, 0, stream>>>(src, dst, dinv, bufA, bufB, E);
    gcn_finalize_kernel<<<grids((long long)N * D, BS), BS, 0, stream>>>(bufB, bufA, dinv, b2, N * D);
    ln(bufB, ln2w, ln2b, N, 128, 1, 0);

    // pre-pool LayerNorm -> H in bufB
    ln(bufB, ppw, ppb, N, 128, 0, 0);

    // gate MLP: 128 -> 128 -> 64 -> 16 -> 1
    cvt(bufB, bfb, (long long)N * D);
    gemm(bfb, g1wb, g1b, bufA, N, 128, 128, 1);
    cvt(bufA, bfb, (long long)N * 128);
    gemm(bfb, g2wb, g2b, bufA, N, 128, 64, 1);
    cvt(bufA, bfb, (long long)N * 64);
    gemm(bfb, g3wb, g3b, bufA, N, 64, 16, 1);
    gate_final_kernel<<<grids(N, BS), BS, 0, stream>>>(bufA, g4W, g4b, gate, N);

    // segment softmax + attention pooling
    zero((float*)gmaxb, G);
    zero(den, G);
    seg_max_kernel<<<grids(N, BS), BS, 0, stream>>>(gate, batch, gmaxb, N);
    seg_exp_kernel<<<grids(N, BS), BS, 0, stream>>>(gate, batch, gmaxb, den, N);
    zero(pooled, (long long)G * D);
    pool_kernel<<<grids((long long)N * 32, BS), BS, 0, stream>>>(bufB, gate, den, batch, pooled, N);

    // MLP head: 128 -> 256(LN,tanh) -> 128(LN,tanh) -> 64
    cvt(pooled, bfs, (long long)G * 128);
    gemm(bfs, m1wb, m1b, z1, G, 128, 256, 0);
    ln(z1, m1lnw, m1lnb, G, 256, 0, 1);
    cvt(z1, bfs, (long long)G * 256);
    gemm(bfs, m2wb, m2b, z2, G, 256, 128, 0);
    ln(z2, m2lnw, m2lnb, G, 128, 0, 1);
    cvt(z2, bfs, (long long)G * 128);
    gemm(bfs, m3wb, m3b, (float*)d_out, G, 128, 64, 0);
}
